// SelfAttention_72627896976028
// MI455X (gfx1250) — matrix operand order repaired and checked by reading, not compiled, not run on hardware
//
#include <hip/hip_runtime.h>

// ---------------------------------------------------------------------------
// CDNA5 (gfx1250, wave32) flash-attention:
//   f,g,h = x@W{f,g,h}+b ; o = softmax(g f^T) h ; out = gamma*o + x
// bf16 WMMA (v_wmma_f32_16x16x32_bf16), fp32 accumulation.
// Attention K/V tiles are streamed into double-buffered LDS by the Tensor
// Data Mover (tensor_load_to_lds, TENSORcnt) so DMA overlaps the WMMAs.
// ---------------------------------------------------------------------------

typedef __bf16 bf16;
typedef __attribute__((ext_vector_type(16))) __bf16 v16bf;
typedef __attribute__((ext_vector_type(8)))  float  v8f;
typedef __attribute__((ext_vector_type(4)))  unsigned int u32x4;
typedef __attribute__((ext_vector_type(8)))  int i32x8;
typedef __attribute__((ext_vector_type(4)))  int i32x4;

#define WMMA_BF16(a, b, c) \
  __builtin_amdgcn_wmma_f32_16x16x32_bf16(false, (a), false, (b), (short)0, (c), false, false)

#define CDIM 128
#define LDIM 4096
#define NROWS 16384  // B*L
#define NKB (LDIM / 64)

static __device__ __forceinline__ unsigned short f2bf(float f) {
  bf16 h = (bf16)f;
  return __builtin_bit_cast(unsigned short, h);
}

// A-matrix fragment, 16x32 bf16, source row-major [16][ld] -> 2x ds_load_b128.
static __device__ __forceinline__ v16bf load_a_rowmajor(const unsigned short* base, int ld,
                                                        int kbase, int lane) {
  const int half = lane >> 4;
  const int m = lane & 15;
  union { unsigned int u[8]; v16bf v; } r;
  const unsigned int* p = (const unsigned int*)(base + m * ld);
#pragma unroll
  for (int t = 0; t < 8; ++t) {
    int c = kbase + ((t & 4) << 2) + half * 8 + ((t & 3) << 1);
    r.u[t] = p[c >> 1];
  }
  return r.v;
}

// B-matrix fragment (32x16): B[k,n] = src[(rowbase+n)*ld + kbase + k]
// (source holds B^T row-major, K contiguous) -> 2x ds_load_b128.
static __device__ __forceinline__ v16bf load_bT(const unsigned short* base, int ld, int rowbase,
                                                int kbase, int lane) {
  const int half = lane >> 4;
  const int n = lane & 15;
  union { unsigned int u[8]; v16bf v; } r;
  const unsigned int* p = (const unsigned int*)(base + (rowbase + n) * ld + kbase);
#pragma unroll
  for (int t = 0; t < 8; ++t) r.u[t] = p[((t & 4) << 1) + half * 4 + (t & 3)];
  return r.v;
}

// ---------------------------------------------------------------------------
// TDM: issue a 2D bf16 tile load (tile_d0 contiguous elems, tile_d1 rows,
// row stride stride0 elems) from global to LDS. D# per CDNA5 ISA ch.8.
// ---------------------------------------------------------------------------
static __device__ __forceinline__ void tdm_load_tile(unsigned lds_byte_off,
                                                     const unsigned short* gptr,
                                                     unsigned tile_d0, unsigned tile_d1,
                                                     unsigned long stride0) {
  unsigned long long ga = (unsigned long long)gptr;
  u32x4 g0;
  g0[0] = 1u;                                   // count=1, user mode, no gather
  g0[1] = lds_byte_off;                         // lds_addr
  g0[2] = (unsigned)ga;                         // global_addr[31:0]
  g0[3] = (unsigned)(ga >> 32) | (2u << 30);    // global_addr[56:32] | type=2
  // tensor_dim0/1 = 1<<20 (no OOB clipping inside our tiles)
  i32x8 g1;
  g1[0] = 1 << 16;                              // workgroup_mask=0, data_size=2B
  g1[1] = 0;                                    // barrier addr=0, tensor_dim0[15:0]=0
  g1[2] = 0x10;                                 // tensor_dim0[31:16]=0x10, tensor_dim1[15:0]=0
  g1[3] = (int)(0x10u | (tile_d0 << 16));       // tensor_dim1[31:16]=0x10, tile_dim0
  g1[4] = (int)(tile_d1 & 0xFFFFu);             // tile_dim1 (tile_dim2=0)
  g1[5] = (int)(stride0 & 0xFFFFFFFFu);         // tensor_dim0_stride[31:0]
  g1[6] = (int)((stride0 >> 32) & 0xFFFFu);     // stride[47:32] | dim1_stride[15:0]=0
  g1[7] = 0;
  i32x4 z4 = {0, 0, 0, 0};
#if defined(__clang_major__) && __clang_major__ >= 23
  i32x8 z8 = {0, 0, 0, 0, 0, 0, 0, 0};
  __builtin_amdgcn_tensor_load_to_lds(g0, g1, z4, z4, z8, 0);
#else
  __builtin_amdgcn_tensor_load_to_lds(g0, g1, z4, z4, 0);
#endif
}

// ---------------------------------------------------------------------------
// Kernel 0: transpose the three 128x128 weight matrices to bf16 W^T[d][c].
// ---------------------------------------------------------------------------
__global__ __launch_bounds__(256) void transpose_w_kernel(
    const float* __restrict__ Wf, const float* __restrict__ Wg,
    const float* __restrict__ Wh, unsigned short* __restrict__ WT) {
  const float* Wp[3] = {Wf, Wg, Wh};
  const int p = blockIdx.x;
  const float* W = Wp[p];
  unsigned short* out = WT + p * CDIM * CDIM;
#pragma unroll
  for (int i = 0; i < 64; ++i) {
    int idx = threadIdx.x + i * 256;  // coalesced read W[c][d]
    int c = idx >> 7, d = idx & 127;
    out[d * CDIM + c] = f2bf(W[idx]);
  }
}

// ---------------------------------------------------------------------------
// Kernel 1: F/G projections (row-major bf16) + H^T (bf16, [b][c][l]).
// ---------------------------------------------------------------------------
__global__ __launch_bounds__(256) void proj_kernel(
    const float* __restrict__ x, const unsigned short* __restrict__ WT,
    const float* __restrict__ bfv, const float* __restrict__ bgv,
    const float* __restrict__ bhv,
    unsigned short* __restrict__ F, unsigned short* __restrict__ G,
    unsigned short* __restrict__ HT) {
  __shared__ unsigned short xs[128 * CDIM];  // 32 KB
  __shared__ unsigned short ws[CDIM * CDIM]; // 32 KB (W^T tile)
  const int tid = threadIdx.x;
  const int lane = tid & 31;
  const int w = tid >> 5;
  const int half = lane >> 4;
  const int m = lane & 15;
  const int row0 = blockIdx.x * 128;

  {
    const float4* xp = (const float4*)(x + (long)row0 * CDIM);
    uint2* xsp = (uint2*)xs;
#pragma unroll
    for (int i = 0; i < 16; ++i) {
      int idx = tid + i * 256;
      float4 v = xp[idx];
      uint2 o;
      o.x = ((unsigned)f2bf(v.y) << 16) | f2bf(v.x);
      o.y = ((unsigned)f2bf(v.w) << 16) | f2bf(v.z);
      xsp[idx] = o;
    }
  }
  __syncthreads();

  v16bf Ax[4];
  {
    const unsigned short* arow = xs + (w * 16) * CDIM;
#pragma unroll
    for (int kc = 0; kc < 4; ++kc) Ax[kc] = load_a_rowmajor(arow, CDIM, kc * 32, lane);
  }

  const float* bp[3] = {bfv, bgv, bhv};

#pragma unroll
  for (int p = 0; p < 3; ++p) {
    __syncthreads();
    {
      const uint4* src = (const uint4*)(WT + p * CDIM * CDIM);
      uint4* dst = (uint4*)ws;
#pragma unroll
      for (int i = 0; i < 8; ++i) dst[tid + i * 256] = src[tid + i * 256];
    }
    __syncthreads();

#pragma unroll
    for (int nt = 0; nt < 8; ++nt) {
      v8f acc = {};
#pragma unroll
      for (int kc = 0; kc < 4; ++kc)
        acc = WMMA_BF16(Ax[kc], load_bT(ws, CDIM, nt * 16, kc * 32, lane), acc);

      const int n = nt * 16 + m;
      const float bias = bp[p][n];
#pragma unroll
      for (int j = 0; j < 8; ++j) {
        long row = row0 + w * 16 + j + 8 * half;  // C/D layout: M = j + 8*half
        unsigned short v = f2bf(acc[j] + bias);
        if (p == 0) {
          F[row * CDIM + n] = v;
        } else if (p == 1) {
          G[row * CDIM + n] = v;
        } else {
          long b = row >> 12;
          long l = row & (LDIM - 1);
          HT[b * (long)CDIM * LDIM + (long)n * LDIM + l] = v;
        }
      }
    }
  }
}

// ---------------------------------------------------------------------------
// Kernel 2: flash attention, TDM double-buffered K/V tiles.
// Block = 4 waves = 64 queries; key blocks of 64.
// ---------------------------------------------------------------------------
__global__ __launch_bounds__(128) void attn_kernel(
    const unsigned short* __restrict__ F, const unsigned short* __restrict__ G,
    const unsigned short* __restrict__ HT, const float* __restrict__ x,
    const float* __restrict__ gamma_p, float* __restrict__ out) {
  __shared__ unsigned short Fs[2][64 * CDIM];   // 2 x 16 KB key tile   [key][c]
  __shared__ unsigned short HsT[2][CDIM * 64];  // 2 x 16 KB value tile [c][key]
  __shared__ unsigned short Ps[4 * 16 * 64];    // 8 KB P staging (per wave)

  const int tid = threadIdx.x;
  const int lane = tid & 31;
  const int w = tid >> 5;
  const int half = lane >> 4;
  const int m = lane & 15;
  const int b = blockIdx.y;
  const int qw = blockIdx.x * 64 + w * 16;
  const long rowbaseB = (long)b * LDIM;

  const unsigned short* Fb = F + rowbaseB * CDIM;
  const unsigned short* HTb = HT + (long)b * CDIM * LDIM;

  const unsigned ldsF[2] = {(unsigned)(unsigned long long)(void*)&Fs[0][0],
                            (unsigned)(unsigned long long)(void*)&Fs[1][0]};
  const unsigned ldsH[2] = {(unsigned)(unsigned long long)(void*)&HsT[0][0],
                            (unsigned)(unsigned long long)(void*)&HsT[1][0]};

  // Q (=G) A-fragments resident in VGPRs.
  v16bf Ga[4];
  {
    const unsigned short* gp = G + (rowbaseB + qw) * CDIM;
#pragma unroll
    for (int kc = 0; kc < 4; ++kc) Ga[kc] = load_a_rowmajor(gp, CDIM, kc * 32, lane);
  }

  float mx[8], sm[8];
  v8f o[8];
#pragma unroll
  for (int j = 0; j < 8; ++j) { mx[j] = -1e30f; sm[j] = 0.0f; }
#pragma unroll
  for (int ct = 0; ct < 8; ++ct) o[ct] = {};

  // Prologue: DMA buffer 0 (F tile is one contiguous 8192-elem row).
  if (w == 0) {
    tdm_load_tile(ldsF[0], Fb, 8192, 1, 8192);
    tdm_load_tile(ldsH[0], HTb, 64, CDIM, LDIM);
  }

  for (int kb = 0; kb < NKB; ++kb) {
    const int cur = kb & 1;
    __syncthreads();  // all waves done reading buffer (kb+1)&1 from iter kb-1
    if (w == 0) {
      if (kb + 1 < NKB) {
        const int nxt = (kb + 1) & 1;
        tdm_load_tile(ldsF[nxt], Fb + (kb + 1) * 64 * CDIM, 8192, 1, 8192);
        tdm_load_tile(ldsH[nxt], HTb + (kb + 1) * 64, 64, CDIM, LDIM);
        __builtin_amdgcn_s_wait_tensorcnt(2);  // in-order: buffer `cur` complete
      } else {
        __builtin_amdgcn_s_wait_tensorcnt(0);
      }
    }
    __syncthreads();  // publish buffer `cur` to all waves

    // ---- S = G * F^T ----
    v8f s[4];
#pragma unroll
    for (int nt = 0; nt < 4; ++nt) s[nt] = {};
#pragma unroll
    for (int kc = 0; kc < 4; ++kc)
#pragma unroll
      for (int nt = 0; nt < 4; ++nt)
        s[nt] = WMMA_BF16(Ga[kc], load_bT(Fs[cur], CDIM, nt * 16, kc * 32, lane), s[nt]);

    // ---- online softmax (row M = j + 8*half across a 16-lane group) ----
#pragma unroll
    for (int j = 0; j < 8; ++j) {
      float bm = -1e30f;
#pragma unroll
      for (int nt = 0; nt < 4; ++nt) {
        float v = s[nt][j];
        v = fmaxf(v, __shfl_xor(v, 1, 32));
        v = fmaxf(v, __shfl_xor(v, 2, 32));
        v = fmaxf(v, __shfl_xor(v, 4, 32));
        v = fmaxf(v, __shfl_xor(v, 8, 32));
        bm = fmaxf(bm, v);
      }
      float nm = fmaxf(mx[j], bm);
      float corr = __expf(mx[j] - nm);
      mx[j] = nm;
      float rs = 0.0f;
#pragma unroll
      for (int nt = 0; nt < 4; ++nt) {
        float p = __expf(s[nt][j] - nm);
        s[nt][j] = p;
        float t = p;
        t += __shfl_xor(t, 1, 32);
        t += __shfl_xor(t, 2, 32);
        t += __shfl_xor(t, 4, 32);
        t += __shfl_xor(t, 8, 32);
        rs += t;
      }
      sm[j] = sm[j] * corr + rs;
#pragma unroll
      for (int ct = 0; ct < 8; ++ct) o[ct][j] = o[ct][j] * corr;
    }

    // ---- P (C/D layout) -> LDS -> A-fragment layout ----
    unsigned short* Pw = Ps + w * 16 * 64;
#pragma unroll
    for (int nt = 0; nt < 4; ++nt)
#pragma unroll
      for (int j = 0; j < 8; ++j)
        Pw[(j + 8 * half) * 64 + nt * 16 + m] = f2bf(s[nt][j]);
    __syncthreads();  // uniform; orders P store->load

    v16bf Pa[2];
#pragma unroll
    for (int kc = 0; kc < 2; ++kc) Pa[kc] = load_a_rowmajor(Pw, 64, kc * 32, lane);

    // ---- O += P * H ----
#pragma unroll
    for (int kc = 0; kc < 2; ++kc)
#pragma unroll
      for (int ct = 0; ct < 8; ++ct)
        o[ct] = WMMA_BF16(Pa[kc], load_bT(HsT[cur], 64, ct * 16, kc * 32, lane), o[ct]);
  }

  // ---- epilogue: out = gamma * (O * 1/rowsum) + x ----
  const float gamma = *gamma_p;
  float inv[8];
#pragma unroll
  for (int j = 0; j < 8; ++j) inv[j] = 1.0f / sm[j];
#pragma unroll
  for (int ct = 0; ct < 8; ++ct) {
    const int c = ct * 16 + m;
#pragma unroll
    for (int j = 0; j < 8; ++j) {
      long row = rowbaseB + qw + j + 8 * half;
      float v = gamma * (o[ct][j] * inv[j]) + x[row * CDIM + c];
      out[row * CDIM + c] = v;
    }
  }
}

// ---------------------------------------------------------------------------
extern "C" void kernel_launch(void* const* d_in, const int* in_sizes, int n_in,
                              void* d_out, int out_size, void* d_ws, size_t ws_size,
                              hipStream_t stream) {
  const float* x   = (const float*)d_in[0];
  const float* Wf  = (const float*)d_in[1];
  const float* bfv = (const float*)d_in[2];
  const float* Wg  = (const float*)d_in[3];
  const float* bgv = (const float*)d_in[4];
  const float* Wh  = (const float*)d_in[5];
  const float* bhv = (const float*)d_in[6];
  const float* gmm = (const float*)d_in[7];

  unsigned short* F  = (unsigned short*)d_ws;       // 4 MB bf16 [row][c]
  unsigned short* G  = F + (size_t)NROWS * CDIM;    // 4 MB bf16 [row][c]
  unsigned short* HT = G + (size_t)NROWS * CDIM;    // 4 MB bf16 [b][c][l]
  unsigned short* WT = HT + (size_t)NROWS * CDIM;   // 96 KB bf16 W^T x3

  transpose_w_kernel<<<3, 256, 0, stream>>>(Wf, Wg, Wh, WT);
  proj_kernel<<<NROWS / 128, 256, 0, stream>>>(x, WT, bfv, bgv, bhv, F, G, HT);

  dim3 grid(LDIM / 64, 4);  // 64 query-blocks x B
  attn_kernel<<<grid, 128, 0, stream>>>(F, G, HT, x, gmm, (float*)d_out);
}
